// WanSelfAttention_53953379173293
// MI455X (gfx1250) — matrix operand order repaired and checked by reading, not compiled, not run on hardware
//
#include <hip/hip_runtime.h>
#include <hip/hip_fp16.h>

typedef __attribute__((ext_vector_type(16))) _Float16 v16h;
typedef __attribute__((ext_vector_type(8)))  _Float16 v8h;
typedef __attribute__((ext_vector_type(8)))  float    v8f;
typedef __attribute__((ext_vector_type(4)))  unsigned int v4u;
typedef __attribute__((ext_vector_type(8)))  int      v8i;
typedef __attribute__((ext_vector_type(4)))  int      v4i;

#define DIMSZ 1536
#define NHEAD 12
#define HDIM  128
#define LSEQ  4096
#define KHALF 64   /* HDIM/2 */

#if __has_builtin(__builtin_amdgcn_tensor_load_to_lds) && __has_builtin(__builtin_amdgcn_s_wait_tensorcnt)
#define HAS_TDM 1
#else
#define HAS_TDM 0
#endif

__device__ __forceinline__ v8f wmma_f16(v16h a, v16h b, v8f c) {
  // D = A(16x32 f16) * B(32x16 f16) + C(16x16 f32)
  return __builtin_amdgcn_wmma_f32_16x16x32_f16(
      /*neg_a=*/false, a, /*neg_b=*/false, b,
      /*c_mod=*/(short)0, c, /*reuse_a=*/false, /*reuse_b=*/false);
}

// Build A-fragment (16x32 f16): lane row m = lane&15; for half=lane>>4,
// elements 0..15 hold k = 16*half + e.
__device__ __forceinline__ v16h load_a_frag(const _Float16* rowp, int hf) {
  v8h lo = *(const v8h*)(rowp + 16 * hf);
  v8h hi = *(const v8h*)(rowp + 16 * hf + 8);
  v16h a;
#pragma unroll
  for (int e = 0; e < 8; ++e) { a[e] = lo[e]; a[8 + e] = hi[e]; }
  return a;
}

#if HAS_TDM
// Issue a TDM 2D tile load (f16 elements): tile_w contiguous elems per row,
// tile_h rows, row stride = stride_elems, into LDS at lds_off (bytes).
// D# bit layout per CDNA5 ISA ch.8 (group0 128b, group1 256b).
__device__ __forceinline__ void tdm_load_2d(unsigned lds_off, const void* gptr,
                                            unsigned tile_w, unsigned tile_h,
                                            unsigned stride_elems) {
  unsigned long long ga = (unsigned long long)gptr;
  v4u g0;
  g0[0] = 1u;                                   // count=1 (valid), user mode
  g0[1] = lds_off;                              // lds_addr [63:32]
  g0[2] = (unsigned)(ga & 0xFFFFFFFFu);         // global_addr lo
  g0[3] = (unsigned)((ga >> 32) & 0x01FFFFFFu)  // global_addr hi (bits 56:32)
          | (2u << 30);                         // type=2 ("image")
  v8i g1;
  g1[0] = (int)(1u << 16);                      // data_size=1 (2 bytes), wg_mask=0
  g1[1] = (int)((tile_w & 0xFFFFu) << 16);      // tensor_dim0 lo16 (== tile_w: no OOB)
  g1[2] = (int)((tile_w >> 16) | ((tile_h & 0xFFFFu) << 16)); // td0 hi / tensor_dim1 lo
  g1[3] = (int)((tile_h >> 16) | (tile_w << 16));             // td1 hi / tile_dim0
  g1[4] = (int)(tile_h & 0xFFFFu);              // tile_dim1, tile_dim2=0
  g1[5] = (int)stride_elems;                    // tensor_dim0_stride lo32
  g1[6] = 0;                                    // stride0 hi16, stride1 lo16
  g1[7] = 0;                                    // stride1 hi
  v4i g2 = {1, 0, 0, 0};                        // tensor_dim2=1, rest 0
  v4i g3 = {0, 0, 0, 0};
#if __clang_major__ >= 23
  v8i g4 = {0, 0, 0, 0, 0, 0, 0, 0};
  __builtin_amdgcn_tensor_load_to_lds(g0, g1, g2, g3, g4, 0);
#else
  __builtin_amdgcn_tensor_load_to_lds(g0, g1, g2, g3, 0);
#endif
}
#endif

// ---------------------------------------------------------------- converts
__global__ void k_cvt_f32_to_f16(const float* __restrict__ src,
                                 _Float16* __restrict__ dst, int n) {
  int i = blockIdx.x * blockDim.x + threadIdx.x;
  if (i < n) dst[i] = (_Float16)src[i];
}

// WT[n][k] = (f16) W[k][n]   (W is [K][N] row-major f32)
__global__ void k_transpose_w(const float* __restrict__ W,
                              _Float16* __restrict__ WT, int K, int N) {
  long i = (long)blockIdx.x * blockDim.x + threadIdx.x;
  if (i < (long)K * N) {
    int k = (int)(i % K);
    int n = (int)(i / K);
    WT[(long)n * K + k] = (_Float16)W[(long)k * N + n];
  }
}

// ---------------------------------------------------------------- GEMM
// One wave computes a 64(M) x 64(N) tile of C = A * BT^T + bias.
// B tiles (64 cols x 32 k, 4KB) are staged in LDS by the Tensor Data Mover,
// double-buffered, one TDM load in flight (TENSORcnt pipelining).
// MODE 0: f16 output scattered into head-split [NH][M][HDIM]; MODE 1: f32 [M][N].
template <int MODE>
__global__ void __launch_bounds__(32)
k_gemm_f16(const _Float16* __restrict__ A,   // [M][K] f16
           const _Float16* __restrict__ BT,  // [N][K] f16 (pre-transposed)
           const float* __restrict__ bias,   // [N]
           _Float16* __restrict__ Oh,        // MODE 0
           float* __restrict__ Of,           // MODE 1
           int M, int N, int K) {
#if HAS_TDM
  __shared__ __align__(64) _Float16 bt[2][64 * 32];   // 2 x 4KB B tiles
#endif
  const int lane = threadIdx.x;
  const int m  = lane & 15;
  const int hf = lane >> 4;
  const int q0 = blockIdx.y * 64;
  const int n0 = blockIdx.x * 64;

  const v8f vz = {0.f, 0.f, 0.f, 0.f, 0.f, 0.f, 0.f, 0.f};
  v8f acc[4][4];   // [qt][j]
#pragma unroll
  for (int qt = 0; qt < 4; ++qt)
#pragma unroll
    for (int j = 0; j < 4; ++j) acc[qt][j] = vz;

  const _Float16* arow[4];
#pragma unroll
  for (int qt = 0; qt < 4; ++qt) arow[qt] = A + (long)(q0 + qt * 16 + m) * K;

  const int nkc = K / 32;
#if HAS_TDM
  const _Float16* btile_g = BT + (long)n0 * K;
  tdm_load_2d((unsigned)(unsigned long long)(&bt[0][0]), btile_g, 32, 64, (unsigned)K);
  for (int kc = 0; kc < nkc; ++kc) {
    int nxt = (kc + 1 == nkc) ? 0 : (kc + 1);   // wrapped dummy prefetch on last iter
    tdm_load_2d((unsigned)(unsigned long long)(&bt[(kc + 1) & 1][0]),
                btile_g + nxt * 32, 32, 64, (unsigned)K);
    __builtin_amdgcn_s_wait_tensorcnt(1);       // chunk kc resident in LDS
    asm volatile("" ::: "memory");
    const _Float16* btile = &bt[kc & 1][0];
    v16h af[4];
#pragma unroll
    for (int qt = 0; qt < 4; ++qt) {
      __builtin_prefetch(arow[qt] + (kc + 2) * 32, 0, 3);
      af[qt] = load_a_frag(arow[qt] + kc * 32, hf);
    }
#pragma unroll
    for (int j = 0; j < 4; ++j) {
      v16h bf = *(const v16h*)(btile + (j * 16 + m) * 32 + 16 * hf);
#pragma unroll
      for (int qt = 0; qt < 4; ++qt) acc[qt][j] = wmma_f16(af[qt], bf, acc[qt][j]);
    }
    asm volatile("" ::: "memory");
  }
#else
  for (int kc = 0; kc < nkc; ++kc) {
    v16h af[4];
#pragma unroll
    for (int qt = 0; qt < 4; ++qt) {
      __builtin_prefetch(arow[qt] + (kc + 2) * 32, 0, 3);
      af[qt] = load_a_frag(arow[qt] + kc * 32, hf);
    }
#pragma unroll
    for (int j = 0; j < 4; ++j) {
      v16h bf = *(const v16h*)(BT + (long)(n0 + j * 16 + m) * K + kc * 32 + 16 * hf);
#pragma unroll
      for (int qt = 0; qt < 4; ++qt) acc[qt][j] = wmma_f16(af[qt], bf, acc[qt][j]);
    }
  }
#endif

#pragma unroll
  for (int qt = 0; qt < 4; ++qt)
#pragma unroll
    for (int j = 0; j < 4; ++j)
#pragma unroll
      for (int r = 0; r < 8; ++r) {
        int row = q0 + qt * 16 + r + 8 * hf;
        int col = n0 + j * 16 + m;
        float v = acc[qt][j][r] + bias[col];
        if (MODE == 0) {
          int h = col >> 7, d = col & 127;   // HDIM = 128
          Oh[((long)h * M + row) * HDIM + d] = (_Float16)v;
        } else {
          Of[(long)row * N + col] = v;
        }
      }
}

// ---------------------------------------------------------------- RoPE (in place on f16 Q,K in [NH][L][HDIM])
__global__ void k_rope_inplace(const float* __restrict__ freqs,
                               _Float16* __restrict__ Q,
                               _Float16* __restrict__ Kb) {
  int idx = blockIdx.x * blockDim.x + threadIdx.x;
  if (idx >= NHEAD * LSEQ) return;
  int l = idx % LSEQ, h = idx / LSEQ;
  _Float16* q = Q  + ((long)h * LSEQ + l) * HDIM;
  _Float16* k = Kb + ((long)h * LSEQ + l) * HDIM;
  const float* fr = freqs + (long)l * KHALF;
#pragma unroll 4
  for (int i = 0; i < KHALF; ++i) {
    float ang = fr[i];
    float c = cosf(ang), s = sinf(ang);
    float q0 = (float)q[2 * i], q1 = (float)q[2 * i + 1];
    q[2 * i]     = (_Float16)(q0 * c - q1 * s);
    q[2 * i + 1] = (_Float16)(q0 * s + q1 * c);
    float k0 = (float)k[2 * i], k1 = (float)k[2 * i + 1];
    k[2 * i]     = (_Float16)(k0 * c - k1 * s);
    k[2 * i + 1] = (_Float16)(k0 * s + k1 * c);
  }
}

// V [NH][L][HDIM] -> Vt [NH][HDIM][L]  (coalesced in l across threads)
__global__ void k_vt(const _Float16* __restrict__ V, _Float16* __restrict__ Vt) {
  int idx = blockIdx.x * blockDim.x + threadIdx.x;
  if (idx >= NHEAD * LSEQ) return;
  int l = idx % LSEQ, h = idx / LSEQ;
  const _Float16* src = V + ((long)h * LSEQ + l) * HDIM;
  _Float16* dst = Vt + (long)h * HDIM * LSEQ + l;
#pragma unroll 8
  for (int d = 0; d < HDIM; ++d) dst[(long)d * LSEQ] = src[d];
}

// ---------------------------------------------------------------- attention
// 4 waves per workgroup; each wave owns a 16-query block (64 queries total).
// K/V 32-key tiles are staged into LDS by the TDM (wave 0), double-buffered,
// and shared by all 4 waves -> 4x less L2 traffic. Flash-style online softmax.
__global__ void __launch_bounds__(128)
k_attn(const _Float16* __restrict__ Q,   // [NH][L][HDIM]
       const _Float16* __restrict__ Km,  // [NH][L][HDIM]
       const _Float16* __restrict__ Vt,  // [NH][HDIM][L]
       _Float16* __restrict__ AO) {      // [L][DIM]
#if HAS_TDM
  __shared__ __align__(64) _Float16 kt[2][32 * HDIM];  // [buf][key][hd]  8KB each
  __shared__ __align__(64) _Float16 vtl[2][HDIM * 32]; // [buf][hd][key]  8KB each
#endif
  __shared__ __align__(32) _Float16 lp[4][16 * 32];    // per-wave P tile [query][key]

  const int lane = threadIdx.x & 31;
  const int wid  = threadIdx.x >> 5;
  const int m  = lane & 15;
  const int hf = lane >> 4;
  const int q0 = blockIdx.x * 64 + wid * 16;
  const int h  = blockIdx.y;
  const float scale = 0.08838834764831845f;  // 1/sqrt(128)

  const _Float16* Qh = Q  + (long)h * LSEQ * HDIM;
  const _Float16* Kh = Km + (long)h * LSEQ * HDIM;
  const _Float16* Vh = Vt + (long)h * HDIM * LSEQ;

  v16h aq[4];
  const _Float16* qrow = Qh + (long)(q0 + m) * HDIM;
#pragma unroll
  for (int kc = 0; kc < 4; ++kc) aq[kc] = load_a_frag(qrow + kc * 32, hf);

  const v8f vz = {0.f, 0.f, 0.f, 0.f, 0.f, 0.f, 0.f, 0.f};
  v8f acc[8] = {vz, vz, vz, vz, vz, vz, vz, vz};
  float mprev[8], ssum[8];
#pragma unroll
  for (int r = 0; r < 8; ++r) { mprev[r] = -3.0e38f; ssum[r] = 0.f; }

  const int nkb = LSEQ / 32;
#if HAS_TDM
  if (wid == 0) {
    tdm_load_2d((unsigned)(unsigned long long)(&kt[0][0]),  Kh, HDIM, 32, HDIM);
    tdm_load_2d((unsigned)(unsigned long long)(&vtl[0][0]), Vh, 32, HDIM, LSEQ);
  }
#endif
  for (int kb = 0; kb < nkb; ++kb) {
    __syncthreads();   // previous iteration's LDS reads complete (WAR)
#if HAS_TDM
    if (wid == 0) {
      int nxt = (kb + 1 == nkb) ? 0 : (kb + 1);  // wrapped dummy prefetch on last iter
      tdm_load_2d((unsigned)(unsigned long long)(&kt[(kb + 1) & 1][0]),
                  Kh + (long)nxt * 32 * HDIM, HDIM, 32, HDIM);
      tdm_load_2d((unsigned)(unsigned long long)(&vtl[(kb + 1) & 1][0]),
                  Vh + nxt * 32, 32, HDIM, LSEQ);
      __builtin_amdgcn_s_wait_tensorcnt(2);      // block kb resident in LDS
    }
    __syncthreads();   // data visible to all waves
    const _Float16* kbuf = &kt[kb & 1][0];       // row stride HDIM (same as global)
    const _Float16* vbuf = &vtl[kb & 1][0];
    const long vstride = 32;
#else
    const _Float16* kbuf = Kh + (long)kb * 32 * HDIM;
    const _Float16* vbuf = Vh + kb * 32;
    const long vstride = LSEQ;
#endif

    v8f s0 = vz, s1 = vz;
#pragma unroll
    for (int kc = 0; kc < 4; ++kc) {
      v16h b0 = *(const v16h*)(kbuf + (long)m        * HDIM + kc * 32 + 16 * hf);
      v16h b1 = *(const v16h*)(kbuf + (long)(16 + m) * HDIM + kc * 32 + 16 * hf);
      s0 = wmma_f16(aq[kc], b0, s0);
      s1 = wmma_f16(aq[kc], b1, s1);
    }

    float p0[8], p1[8], alpha[8];
#pragma unroll
    for (int r = 0; r < 8; ++r) {
      float a = s0[r] * scale, b = s1[r] * scale;
      float mx = fmaxf(a, b);
#pragma unroll
      for (int d = 1; d < 16; d <<= 1) mx = fmaxf(mx, __shfl_xor(mx, d, 16));
      float mnew = fmaxf(mprev[r], mx);
      alpha[r] = __expf(mprev[r] - mnew);
      p0[r] = __expf(a - mnew);
      p1[r] = __expf(b - mnew);
      float rs = p0[r] + p1[r];
#pragma unroll
      for (int d = 1; d < 16; d <<= 1) rs += __shfl_xor(rs, d, 16);
      ssum[r] = ssum[r] * alpha[r] + rs;
      mprev[r] = mnew;
    }

    _Float16* lpw = &lp[wid][0];
#pragma unroll
    for (int r = 0; r < 8; ++r) {
      int qr = r + 8 * hf;
      lpw[qr * 32 + m]      = (_Float16)p0[r];
      lpw[qr * 32 + 16 + m] = (_Float16)p1[r];
    }
    v8h plo = *(const v8h*)(&lpw[m * 32 + 16 * hf]);
    v8h phi = *(const v8h*)(&lpw[m * 32 + 16 * hf + 8]);
    v16h ap;
#pragma unroll
    for (int e = 0; e < 8; ++e) { ap[e] = plo[e]; ap[8 + e] = phi[e]; }

#pragma unroll
    for (int j = 0; j < 8; ++j)
#pragma unroll
      for (int r = 0; r < 8; ++r) acc[j][r] *= alpha[r];

#pragma unroll
    for (int j = 0; j < 8; ++j) {
      v16h bv = *(const v16h*)(vbuf + (long)(j * 16 + m) * vstride + 16 * hf);
      acc[j] = wmma_f16(ap, bv, acc[j]);
    }
  }

#pragma unroll
  for (int r = 0; r < 8; ++r) {
    float inv = 1.0f / ssum[r];
    int row = q0 + r + 8 * hf;
#pragma unroll
    for (int j = 0; j < 8; ++j)
      AO[(long)row * DIMSZ + h * HDIM + j * 16 + m] = (_Float16)(acc[j][r] * inv);
  }
}

// ---------------------------------------------------------------- launch
extern "C" void kernel_launch(void* const* d_in, const int* in_sizes, int n_in,
                              void* d_out, int out_size, void* d_ws, size_t ws_size,
                              hipStream_t stream) {
  const float* x  = (const float*)d_in[0];
  const float* fr = (const float*)d_in[1];
  const float* Wq = (const float*)d_in[2];
  const float* bq = (const float*)d_in[3];
  const float* Wk = (const float*)d_in[4];
  const float* bk = (const float*)d_in[5];
  const float* Wv = (const float*)d_in[6];
  const float* bv = (const float*)d_in[7];
  const float* Wo = (const float*)d_in[8];
  const float* bo = (const float*)d_in[9];
  float* out = (float*)d_out;

  char* ws = (char*)d_ws;
  size_t off = 0;
  auto carve = [&](size_t bytes) -> char* {
    char* p = ws + off;
    off += (bytes + 255) & ~(size_t)255;
    return p;
  };
  const size_t XB = (size_t)LSEQ * DIMSZ * sizeof(_Float16);
  const size_t WB = (size_t)DIMSZ * DIMSZ * sizeof(_Float16);
  _Float16* xh  = (_Float16*)carve(XB);
  _Float16* WqT = (_Float16*)carve(WB);
  _Float16* WkT = (_Float16*)carve(WB);
  _Float16* WvT = (_Float16*)carve(WB);
  _Float16* WoT = (_Float16*)carve(WB);
  _Float16* Qb  = (_Float16*)carve(XB);  // [NH][L][HDIM]
  _Float16* Kb  = (_Float16*)carve(XB);
  _Float16* Vb  = (_Float16*)carve(XB);
  _Float16* Vtb = (_Float16*)carve(XB);  // [NH][HDIM][L]
  _Float16* AO  = (_Float16*)carve(XB);  // [L][DIM]

  const int n  = LSEQ * DIMSZ;
  const int nw = DIMSZ * DIMSZ;
  k_cvt_f32_to_f16<<<(n + 255) / 256, 256, 0, stream>>>(x, xh, n);
  k_transpose_w<<<(nw + 255) / 256, 256, 0, stream>>>(Wq, WqT, DIMSZ, DIMSZ);
  k_transpose_w<<<(nw + 255) / 256, 256, 0, stream>>>(Wk, WkT, DIMSZ, DIMSZ);
  k_transpose_w<<<(nw + 255) / 256, 256, 0, stream>>>(Wv, WvT, DIMSZ, DIMSZ);
  k_transpose_w<<<(nw + 255) / 256, 256, 0, stream>>>(Wo, WoT, DIMSZ, DIMSZ);

  dim3 ggrid(DIMSZ / 64, LSEQ / 64);
  k_gemm_f16<0><<<ggrid, 32, 0, stream>>>(xh, WqT, bq, Qb, nullptr, LSEQ, DIMSZ, DIMSZ);
  k_gemm_f16<0><<<ggrid, 32, 0, stream>>>(xh, WkT, bk, Kb, nullptr, LSEQ, DIMSZ, DIMSZ);
  k_gemm_f16<0><<<ggrid, 32, 0, stream>>>(xh, WvT, bv, Vb, nullptr, LSEQ, DIMSZ, DIMSZ);

  const int nhl = NHEAD * LSEQ;
  k_rope_inplace<<<(nhl + 255) / 256, 256, 0, stream>>>(fr, Qb, Kb);
  k_vt<<<(nhl + 255) / 256, 256, 0, stream>>>(Vb, Vtb);

  dim3 agrid(LSEQ / 64, NHEAD);
  k_attn<<<agrid, 128, 0, stream>>>(Qb, Kb, Vtb, AO);

  k_gemm_f16<1><<<ggrid, 32, 0, stream>>>(AO, WoT, bo, nullptr, out, LSEQ, DIMSZ, DIMSZ);
}
